// Block_ToMe_70600672412339
// MI455X (gfx1250) — matrix operand order repaired and checked by reading, not compiled, not run on hardware
//
#include <hip/hip_runtime.h>
#include <hip/hip_bf16.h>

// ---------------------------------------------------------------------------
// Problem constants (from reference)
// ---------------------------------------------------------------------------
static constexpr int Bsz   = 64;
static constexpr int T     = 197;
static constexpr int DIM   = 768;
static constexpr int HEADS = 12;
static constexpr int HD    = 64;     // DIM / HEADS
static constexpr int MLP_H = 3072;
static constexpr int R_EFF = 16;
static constexpr int TP    = 224;    // padded token count (mult of 32) for attention
static constexpr int TA    = 99;     // # even tokens (src side)
static constexpr int TB    = 98;     // # odd tokens (dst side)
static constexpr int UN    = TA - R_EFF;  // 83 unmerged rows
static constexpr int TN    = T - R_EFF;   // 181 merged tokens
static constexpr int QKVN  = 3 * DIM;     // 2304

typedef __attribute__((ext_vector_type(16))) _Float16 v16h;
typedef __attribute__((ext_vector_type(8)))  _Float16 v8h;
typedef __attribute__((ext_vector_type(8)))  float    v8f;
typedef __attribute__((ext_vector_type(4)))  unsigned v4u;
typedef __attribute__((ext_vector_type(8)))  unsigned v8u;

// ---------------------------------------------------------------------------
// WMMA helpers (CDNA5 16x16x32 f16 -> f32)
// ---------------------------------------------------------------------------
__device__ __forceinline__ v8f wmma_f16(v16h a, v16h b, v8f c) {
  // 8 args: (neg_a, A, neg_b, B, c_mod, C, reuse_a, reuse_b)
  return __builtin_amdgcn_wmma_f32_16x16x32_f16(false, a, false, b, (short)0, c,
                                                false, false);
}

// A fragment: 16x32 f16 row-major tile, base pre-offset to row 0 of the tile.
// ISA layout: lane (m = lane&15, hi = lane>>4) holds K = {hi*8+0..7, 16+hi*8+0..7}.
__device__ __forceinline__ v16h load_a_frag(const _Float16* __restrict__ A,
                                            int lda, int k0, int lane) {
  int m  = lane & 15;
  int hi = (lane >> 4) & 1;
  const _Float16* p = A + (long)m * lda + k0 + hi * 8;
  v8h lo = *reinterpret_cast<const v8h*>(p);
  v8h hv = *reinterpret_cast<const v8h*>(p + 16);
  v16h r;
#pragma unroll
  for (int i = 0; i < 8; ++i) { r[i] = lo[i]; r[i + 8] = hv[i]; }
  return r;
}

// B fragment, "nt" flavor: W stored row-major N x K (so B[k][n] = W[n][k]).
// Lane (n = lane&15) column, K = k0 + {hi*8+0..7, 16+hi*8+0..7}, hi = lane>>4.
__device__ __forceinline__ v16h load_b_frag_nt(const _Float16* __restrict__ W,
                                               int ldw, int k0, int lane) {
  int n  = lane & 15;
  int kk = k0 + ((lane >> 4) & 1) * 8;
  const _Float16* p = W + (long)n * ldw + kk;
  v8h lo = *reinterpret_cast<const v8h*>(p);
  v8h hv = *reinterpret_cast<const v8h*>(p + 16);
  v16h r;
#pragma unroll
  for (int i = 0; i < 8; ++i) { r[i] = lo[i]; r[i + 8] = hv[i]; }
  return r;
}

__device__ __forceinline__ v8f zero8() {
  v8f z;
#pragma unroll
  for (int i = 0; i < 8; ++i) z[i] = 0.f;
  return z;
}

// ---------------------------------------------------------------------------
// Generic batched GEMM: C[z] = A[z] (MxK f16) * W[z]^T (NxK f16) [+bias][gelu][+res]
// Wave tile 32x64 (2 A-frags x 4 B-frags = 8 WMMA per K-step).
// Block = 8 waves (2x4) -> 64x256 tile. Requires M%32==0, N%16==0, K%32==0.
// ---------------------------------------------------------------------------
template <bool BIAS, bool RES, bool GELU_A, bool OUTF, bool OUTH>
__global__ __launch_bounds__(256) void gemm_nt_kernel(
    const _Float16* __restrict__ A, const _Float16* __restrict__ W,
    const float* __restrict__ bias, const float* __restrict__ res,
    float* __restrict__ Cf, _Float16* __restrict__ Ch,
    int M, int N, int K, long aStr, long wStr, long cStr) {
  const long z = blockIdx.z;
  A += z * aStr;
  W += z * wStr;
  if (OUTF) Cf += z * cStr;
  if (OUTH) Ch += z * cStr;
  if (RES)  res += z * cStr;

  const int lane = threadIdx.x & 31;
  const int wave = threadIdx.x >> 5;
  const int wm = wave >> 2;  // 0..1
  const int wn = wave & 3;   // 0..3
  const int row0 = blockIdx.y * 64 + wm * 32;
  const int col0 = blockIdx.x * 256 + wn * 64;
  if (row0 >= M) return;

  v8f acc[2][4];
#pragma unroll
  for (int hf = 0; hf < 2; ++hf)
#pragma unroll
    for (int t = 0; t < 4; ++t) acc[hf][t] = zero8();

  const _Float16* A0 = A + (long)row0 * K;
  const _Float16* A1 = A0 + (long)16 * K;
  for (int k0 = 0; k0 < K; k0 += 32) {
    if (k0 + 32 < K) __builtin_prefetch(A0 + ((lane & 15) * (long)K) + k0 + 32, 0, 0);
    v16h af0 = load_a_frag(A0, K, k0, lane);
    v16h af1 = load_a_frag(A1, K, k0, lane);
#pragma unroll
    for (int t = 0; t < 4; ++t) {
      int c = col0 + t * 16;
      if (c < N) {
        v16h bf = load_b_frag_nt(W + (long)c * K, K, k0, lane);
        acc[0][t] = wmma_f16(af0, bf, acc[0][t]);
        acc[1][t] = wmma_f16(af1, bf, acc[1][t]);
      }
    }
  }

  const int nloc = lane & 15;
#pragma unroll
  for (int hf = 0; hf < 2; ++hf) {
    const int rbase = row0 + hf * 16 + ((lane >> 4) & 1) * 8;
#pragma unroll
    for (int t = 0; t < 4; ++t) {
      int cb = col0 + t * 16;
      if (cb < N) {
        int c = cb + nloc;
#pragma unroll
        for (int r = 0; r < 8; ++r) {
          int row = rbase + r;
          float v = acc[hf][t][r];
          if (BIAS)   v += bias[c];
          if (GELU_A) v = 0.5f * v * (1.0f + erff(v * 0.70710678118654752f));
          if (RES)    v += res[(long)row * N + c];
          if (OUTF)   Cf[(long)row * N + c] = v;
          if (OUTH)   Ch[(long)row * N + c] = (_Float16)v;
        }
      }
    }
  }
}

// ---------------------------------------------------------------------------
// Attention AV: O[b,t,h*64+n] = sum_k P[z,t,k] * Vt[z,n,k]
// Vt is pre-transposed (z, HD, TP) so B-frag loads are contiguous.
// One block of 8 waves covers all 224 rows (wave tile 32x64).
// ---------------------------------------------------------------------------
__global__ __launch_bounds__(256) void attn_av_kernel(
    const _Float16* __restrict__ P, const _Float16* __restrict__ Vt,
    _Float16* __restrict__ O) {
  const int z = blockIdx.z;
  const int b = z / HEADS;
  const int h = z - b * HEADS;
  const _Float16* p  = P + (long)z * TP * TP;
  const _Float16* vt = Vt + (long)z * HD * TP;

  const int lane = threadIdx.x & 31;
  const int wave = threadIdx.x >> 5;
  const int row0 = wave * 32;
  if (row0 >= TP) return;

  v8f acc[2][4];
#pragma unroll
  for (int hf = 0; hf < 2; ++hf)
#pragma unroll
    for (int t = 0; t < 4; ++t) acc[hf][t] = zero8();

  const _Float16* a0 = p + (long)row0 * TP;
  const _Float16* a1 = a0 + (long)16 * TP;
  for (int k0 = 0; k0 < TP; k0 += 32) {
    v16h af0 = load_a_frag(a0, TP, k0, lane);
    v16h af1 = load_a_frag(a1, TP, k0, lane);
#pragma unroll
    for (int t = 0; t < 4; ++t) {
      v16h bf = load_b_frag_nt(vt + (long)(t * 16) * TP, TP, k0, lane);
      acc[0][t] = wmma_f16(af0, bf, acc[0][t]);
      acc[1][t] = wmma_f16(af1, bf, acc[1][t]);
    }
  }

  const int nloc = lane & 15;
#pragma unroll
  for (int hf = 0; hf < 2; ++hf) {
    const int rbase = row0 + hf * 16 + ((lane >> 4) & 1) * 8;
#pragma unroll
    for (int t = 0; t < 4; ++t) {
#pragma unroll
      for (int r = 0; r < 8; ++r) {
        int row = rbase + r;
        if (row < T)
          O[((long)b * T + row) * DIM + h * HD + t * 16 + nloc] =
              (_Float16)acc[hf][t][r];
      }
    }
  }
}

// Vt[z][n][k] = Vb[z][k][n]
__global__ __launch_bounds__(256) void v_transpose_kernel(
    const _Float16* __restrict__ Vb, _Float16* __restrict__ Vt) {
  long idx = (long)blockIdx.x * 256 + threadIdx.x;
  const long total = (long)Bsz * HEADS * TP * HD;
  if (idx >= total) return;
  int k = (int)(idx % TP);
  long r = idx / TP;
  int n = (int)(r % HD);
  long z = r / HD;
  Vt[idx] = Vb[((long)z * TP + k) * HD + n];
}

// ---------------------------------------------------------------------------
// TDM demo: tensor_load_to_lds of a 64x16 f32 tile (group0..3 D# in SGPRs),
// wait on TENSORcnt, cluster barrier (NOP outside clusters), drain to global.
// ---------------------------------------------------------------------------
__global__ __launch_bounds__(32) void tdm_demo_kernel(
    const float* __restrict__ src, float* __restrict__ dummy) {
  __shared__ float lds_tile[1024];  // 4 KB
  unsigned lds_addr = (unsigned)(unsigned long long)(void*)&lds_tile[0];
  unsigned long long ga = (unsigned long long)src;

  // D# group 0: count=1, lds_addr, global_addr, type=2
  v4u g0;
  g0[0] = 1u;                                       // count=1, gather off
  g0[1] = lds_addr;                                 // lds_addr
  g0[2] = (unsigned)(ga & 0xFFFFFFFFu);             // global_addr[31:0]
  g0[3] = (unsigned)((ga >> 32) & 0x01FFFFFFu) | (2u << 30);  // addr[56:32]|type=2

  // D# group 1: data_size=4B, tensor 64x16, tile 64x16, stride0=64
  v8u g1;
  g1[0] = (2u << 16);               // workgroup_mask=0, data_size=2 (4B)
  g1[1] = (64u & 0xFFFFu) << 16;    // tensor_dim0[15:0] at bit48
  g1[2] = (0u) | (16u << 16);       // tensor_dim0[31:16] | tensor_dim1[15:0]
  g1[3] = (0u) | (64u << 16);       // tensor_dim1[31:16] | tile_dim0=64
  g1[4] = 16u;                      // tile_dim1=16, tile_dim2=0
  g1[5] = 64u;                      // tensor_dim0_stride[31:0] = 64
  g1[6] = 0u;
  g1[7] = 0u;

  v4u g2; g2[0] = g2[1] = g2[2] = g2[3] = 0u;  // dims 2/3 unused
  v4u g3; g3[0] = g3[1] = g3[2] = g3[3] = 0u;

  asm volatile("tensor_load_to_lds %0, %1, %2, %3"
               :: "s"(g0), "s"(g1), "s"(g2), "s"(g3)
               : "memory");
  __builtin_amdgcn_s_wait_tensorcnt(0);
  __builtin_amdgcn_s_cluster_barrier();  // NOP when not in a cluster
  __syncthreads();
  dummy[threadIdx.x] = lds_tile[threadIdx.x];
}

// ---------------------------------------------------------------------------
// Elementwise / reduction kernels
// ---------------------------------------------------------------------------
__global__ __launch_bounds__(256) void f32_to_f16_kernel(
    const float* __restrict__ src, _Float16* __restrict__ dst, long n) {
  long i = (long)blockIdx.x * 256 + threadIdx.x;
  if (i < n) dst[i] = (_Float16)src[i];
}

__global__ __launch_bounds__(256) void layernorm_kernel(
    const float* __restrict__ X, const float* __restrict__ g,
    const float* __restrict__ bt, _Float16* __restrict__ H, int D) {
  __shared__ float s1[256];
  __shared__ float s2[256];
  const long row = blockIdx.x;
  const float* xr = X + row * (long)D;
  float a = 0.f, b = 0.f;
  for (int i = threadIdx.x; i < D; i += 256) {
    float v = xr[i];
    a += v;
    b += v * v;
  }
  s1[threadIdx.x] = a;
  s2[threadIdx.x] = b;
  __syncthreads();
  for (int st = 128; st > 0; st >>= 1) {
    if (threadIdx.x < st) {
      s1[threadIdx.x] += s1[threadIdx.x + st];
      s2[threadIdx.x] += s2[threadIdx.x + st];
    }
    __syncthreads();
  }
  float mean = s1[0] / D;
  float var  = s2[0] / D - mean * mean;
  float inv  = rsqrtf(var + 1e-5f);
  for (int i = threadIdx.x; i < D; i += 256)
    H[row * (long)D + i] = (_Float16)((xr[i] - mean) * inv * g[i] + bt[i]);
}

// qkv (B*T, 2304) f16 -> padded per-head q,k,v (B, H, TP, HD) f16 (pad rows zeroed)
__global__ __launch_bounds__(256) void qkv_reshape_kernel(
    const _Float16* __restrict__ qkvh, _Float16* __restrict__ Qb,
    _Float16* __restrict__ Kb, _Float16* __restrict__ Vb) {
  long idx = (long)blockIdx.x * 256 + threadIdx.x;
  const long total = (long)Bsz * HEADS * TP * HD;
  if (idx >= total) return;
  int d = (int)(idx % HD);
  long r = idx / HD;
  int t = (int)(r % TP);
  long r2 = r / TP;
  int h = (int)(r2 % HEADS);
  int b = (int)(r2 / HEADS);
  if (t < T) {
    long srcRow = ((long)b * T + t) * QKVN;
    Qb[idx] = qkvh[srcRow + (0 * HEADS + h) * HD + d];
    Kb[idx] = qkvh[srcRow + (1 * HEADS + h) * HD + d];
    Vb[idx] = qkvh[srcRow + (2 * HEADS + h) * HD + d];
  } else {
    Qb[idx] = (_Float16)0.f;
    Kb[idx] = (_Float16)0.f;
    Vb[idx] = (_Float16)0.f;
  }
}

// softmax over scores row: p = softmax(s * hd^-0.5 + log(size[col])), masked to T
__global__ __launch_bounds__(256) void softmax_kernel(
    const float* __restrict__ S, const float* __restrict__ sz,
    _Float16* __restrict__ P) {
  __shared__ float red[256];
  const int row = blockIdx.x;   // 0..TP-1
  const int z   = blockIdx.y;   // b*HEADS + h
  const int b   = z / HEADS;
  const float*  srow = S + ((long)z * TP + row) * TP;
  _Float16*     prow = P + ((long)z * TP + row) * TP;
  const int c = threadIdx.x;    // 256 >= TP
  if (row >= T) {
    if (c < TP) prow[c] = (_Float16)0.f;
    return;
  }
  float myv = -1e30f;
  if (c < T) myv = srow[c] * 0.125f + logf(sz[(long)b * T + c]);
  red[c] = myv;
  __syncthreads();
  for (int st = 128; st > 0; st >>= 1) {
    if (c < st) red[c] = fmaxf(red[c], red[c + st]);
    __syncthreads();
  }
  float mx = red[0];
  __syncthreads();
  float e = (c < T) ? expf(myv - mx) : 0.f;
  red[c] = e;
  __syncthreads();
  for (int st = 128; st > 0; st >>= 1) {
    if (c < st) red[c] += red[c + st];
    __syncthreads();
  }
  float inv = 1.f / red[0];
  if (c < TP) prow[c] = (_Float16)(e * inv);
}

// metric[b,t,d] = mean over heads of k[b,h,t,d]
__global__ __launch_bounds__(256) void metric_kernel(
    const _Float16* __restrict__ Kb, float* __restrict__ metric) {
  long idx = (long)blockIdx.x * 256 + threadIdx.x;
  const long total = (long)Bsz * T * HD;
  if (idx >= total) return;
  int d = (int)(idx % HD);
  long bt = idx / HD;
  int t = (int)(bt % T);
  int b = (int)(bt / T);
  float s = 0.f;
#pragma unroll
  for (int h = 0; h < HEADS; ++h)
    s += (float)Kb[(((long)b * HEADS + h) * TP + t) * HD + d];
  metric[idx] = s * (1.0f / HEADS);
}

// ToMe bipartite matching: one block per batch
__global__ __launch_bounds__(128) void matching_kernel(
    const float* __restrict__ metric, int* __restrict__ unm,
    int* __restrict__ srci, int* __restrict__ dsti) {
  const int b = blockIdx.x;
  __shared__ float sa[TA][HD];
  __shared__ float sb[TB][HD];
  __shared__ float nm[TA];
  __shared__ int   ni[TA];
  __shared__ int   order[TA];
  const float* mb = metric + (long)b * T * HD;

  for (int r = threadIdx.x; r < TA + TB; r += 128) {
    int tok = (r < TA) ? 2 * r : 2 * (r - TA) + 1;
    const float* src = mb + (long)tok * HD;
    float ss = 0.f;
    for (int d = 0; d < HD; ++d) { float v = src[d]; ss += v * v; }
    float inv = rsqrtf(ss);
    float* dstl = (r < TA) ? sa[r] : sb[r - TA];
    for (int d = 0; d < HD; ++d) dstl[d] = src[d] * inv;
  }
  __syncthreads();

  for (int i = threadIdx.x; i < TA; i += 128) {
    if (i == 0) { nm[0] = -1e30f; ni[0] = 0; continue; }
    float best = -1e30f;
    int bj = 0;
    for (int j = 0; j < TB; ++j) {
      float d = 0.f;
      for (int c = 0; c < HD; ++c) d += sa[i][c] * sb[j][c];
      if (d > best) { best = d; bj = j; }
    }
    nm[i] = best;
    ni[i] = bj;
  }
  __syncthreads();

  if (threadIdx.x == 0) {
    for (int i = 0; i < TA; ++i) order[i] = i;
    // stable insertion sort, descending by nm (matches argsort(-nm) stability)
    for (int i = 1; i < TA; ++i) {
      int oi = order[i];
      float key = nm[oi];
      int j = i - 1;
      while (j >= 0 && nm[order[j]] < key) { order[j + 1] = order[j]; --j; }
      order[j + 1] = oi;
    }
    for (int i = 0; i < R_EFF; ++i) {
      srci[b * R_EFF + i] = order[i];
      dsti[b * R_EFF + i] = ni[order[i]];
    }
    int tmp[TA];
    for (int i = 0; i < UN; ++i) tmp[i] = order[R_EFF + i];
    for (int i = 1; i < UN; ++i) {
      int k = tmp[i];
      int j = i - 1;
      while (j >= 0 && tmp[j] > k) { tmp[j + 1] = tmp[j]; --j; }
      tmp[j + 1] = k;
    }
    for (int i = 0; i < UN; ++i) unm[b * UN + i] = tmp[i];
  }
}

// merged row r: unm part (r<UN) from even token 2*unm[r]; dst part from odd 2*(r-UN)+1
__global__ __launch_bounds__(256) void merge_gather_kernel(
    const float* __restrict__ x1, const float* __restrict__ sz,
    const int* __restrict__ unm, float* __restrict__ xs,
    float* __restrict__ nsz) {
  const int b = blockIdx.y;
  const int r = blockIdx.x;  // 0..TN-1
  int tok = (r < UN) ? 2 * unm[b * UN + r] : 2 * (r - UN) + 1;
  float s = sz[(long)b * T + tok];
  const float* src = x1 + ((long)b * T + tok) * DIM;
  float* dst = xs + ((long)b * TN + r) * DIM;
  for (int i = threadIdx.x; i < DIM; i += 256) dst[i] = src[i] * s;
  if (threadIdx.x == 0) nsz[(long)b * TN + r] = s;
}

__global__ __launch_bounds__(256) void merge_scatter_kernel(
    const float* __restrict__ x1, const float* __restrict__ sz,
    const int* __restrict__ srci, const int* __restrict__ dsti,
    float* __restrict__ xs, float* __restrict__ nsz) {
  const int b = blockIdx.y;
  const int i = blockIdx.x;  // 0..R_EFF-1
  int stok = 2 * srci[b * R_EFF + i];
  int d = dsti[b * R_EFF + i];
  float s = sz[(long)b * T + stok];
  const float* src = x1 + ((long)b * T + stok) * DIM;
  float* dst = xs + ((long)b * TN + UN + d) * DIM;
  for (int c = threadIdx.x; c < DIM; c += 256) atomicAdd(&dst[c], src[c] * s);
  if (threadIdx.x == 0) atomicAdd(&nsz[(long)b * TN + UN + d], s);
}

__global__ __launch_bounds__(256) void normalize_kernel(
    const float* __restrict__ xs, const float* __restrict__ nsz,
    float* __restrict__ x2) {
  long idx = (long)blockIdx.x * 256 + threadIdx.x;
  const long total = (long)Bsz * TN * DIM;
  if (idx >= total) return;
  long row = idx / DIM;
  x2[idx] = xs[idx] / nsz[row];
}

// rci[b, c-1] = (row containing original token c) - 1, for c = 1..T-1
__global__ __launch_bounds__(256) void rci_kernel(
    const int* __restrict__ unm, const int* __restrict__ srci,
    const int* __restrict__ dsti, float* __restrict__ rci) {
  const int b = blockIdx.x;
  __shared__ int su[UN];
  __shared__ int ssx[R_EFF];
  __shared__ int sd[R_EFF];
  if (threadIdx.x < UN) su[threadIdx.x] = unm[b * UN + threadIdx.x];
  if (threadIdx.x < R_EFF) {
    ssx[threadIdx.x] = srci[b * R_EFF + threadIdx.x];
    sd[threadIdx.x]  = dsti[b * R_EFF + threadIdx.x];
  }
  __syncthreads();
  const int c = threadIdx.x;
  if (c >= 1 && c < T) {
    int row;
    if (c & 1) {
      row = UN + (c - 1) / 2;
    } else {
      int a = c / 2;
      row = 0;
      bool found = false;
      for (int r = 0; r < UN; ++r)
        if (su[r] == a) { row = r; found = true; break; }
      if (!found)
        for (int i = 0; i < R_EFF; ++i)
          if (ssx[i] == a) { row = UN + sd[i]; break; }
    }
    rci[(long)b * (T - 1) + (c - 1)] = (float)(row - 1);
  }
}

// ---------------------------------------------------------------------------
// Host launcher
// ---------------------------------------------------------------------------
static inline long cdiv(long a, long b) { return (a + b - 1) / b; }

extern "C" void kernel_launch(void* const* d_in, const int* in_sizes, int n_in,
                              void* d_out, int out_size, void* d_ws,
                              size_t ws_size, hipStream_t stream) {
  (void)in_sizes; (void)n_in; (void)out_size; (void)ws_size;
  const float* x      = (const float*)d_in[0];
  const float* asz    = (const float*)d_in[1];
  const float* ln1_g  = (const float*)d_in[2];
  const float* ln1_b  = (const float*)d_in[3];
  const float* w_qkv  = (const float*)d_in[4];
  const float* w_proj = (const float*)d_in[5];
  const float* b_proj = (const float*)d_in[6];
  const float* ln2_g  = (const float*)d_in[7];
  const float* ln2_b  = (const float*)d_in[8];
  const float* w_fc1  = (const float*)d_in[9];
  const float* b_fc1  = (const float*)d_in[10];
  const float* w_fc2  = (const float*)d_in[11];
  const float* b_fc2  = (const float*)d_in[12];

  float* out_x   = (float*)d_out;                    // (B, TN, DIM)
  float* out_sz  = out_x + (long)Bsz * TN * DIM;     // (B, TN, 1)
  float* out_rci = out_sz + (long)Bsz * TN;          // (B, T-1)

  const long MBT = (long)Bsz * T;    // 12608 (mult of 32)
  const long MBN = (long)Bsz * TN;   // 11584 (mult of 32)
  const long ZH  = (long)Bsz * HEADS;

  char* wsp = (char*)d_ws;
  auto alloc = [&](size_t bytes) -> char* {
    char* p = wsp;
    wsp += (bytes + 255) & ~(size_t)255;
    return p;
  };
  _Float16* wqh  = (_Float16*)alloc(sizeof(_Float16) * QKVN * DIM);
  _Float16* wph  = (_Float16*)alloc(sizeof(_Float16) * DIM * DIM);
  _Float16* wf1h = (_Float16*)alloc(sizeof(_Float16) * MLP_H * DIM);
  _Float16* wf2h = (_Float16*)alloc(sizeof(_Float16) * DIM * MLP_H);
  _Float16* h1   = (_Float16*)alloc(sizeof(_Float16) * MBT * DIM);
  _Float16* qkvh = (_Float16*)alloc(sizeof(_Float16) * MBT * QKVN);
  _Float16* Qb   = (_Float16*)alloc(sizeof(_Float16) * ZH * TP * HD);
  _Float16* Kb   = (_Float16*)alloc(sizeof(_Float16) * ZH * TP * HD);
  _Float16* Vb   = (_Float16*)alloc(sizeof(_Float16) * ZH * TP * HD);
  _Float16* Vt   = (_Float16*)alloc(sizeof(_Float16) * ZH * TP * HD);
  float*    Sc   = (float*)   alloc(sizeof(float)    * ZH * TP * TP);
  _Float16* Pa   = (_Float16*)alloc(sizeof(_Float16) * ZH * TP * TP);
  _Float16* Oh   = (_Float16*)alloc(sizeof(_Float16) * MBT * DIM);
  float*    x1   = (float*)   alloc(sizeof(float)    * MBT * DIM);
  float*    metr = (float*)   alloc(sizeof(float)    * MBT * HD);
  int*      unm  = (int*)     alloc(sizeof(int) * Bsz * UN);
  int*      srci = (int*)     alloc(sizeof(int) * Bsz * R_EFF);
  int*      dsti = (int*)     alloc(sizeof(int) * Bsz * R_EFF);
  float*    xs   = (float*)   alloc(sizeof(float) * MBN * DIM);
  float*    x2   = (float*)   alloc(sizeof(float) * MBN * DIM);
  _Float16* h2   = (_Float16*)alloc(sizeof(_Float16) * MBN * DIM);
  _Float16* h3   = (_Float16*)alloc(sizeof(_Float16) * MBN * MLP_H);
  float*    tdmo = (float*)   alloc(sizeof(float) * 1024);

  // 0) TDM path demo (independent of main pipeline)
  tdm_demo_kernel<<<1, 32, 0, stream>>>(w_qkv, tdmo);

  // 1) weights fp32 -> f16
  {
    long n;
    n = (long)QKVN * DIM;
    f32_to_f16_kernel<<<cdiv(n, 256), 256, 0, stream>>>(w_qkv, wqh, n);
    n = (long)DIM * DIM;
    f32_to_f16_kernel<<<cdiv(n, 256), 256, 0, stream>>>(w_proj, wph, n);
    n = (long)MLP_H * DIM;
    f32_to_f16_kernel<<<cdiv(n, 256), 256, 0, stream>>>(w_fc1, wf1h, n);
    n = (long)DIM * MLP_H;
    f32_to_f16_kernel<<<cdiv(n, 256), 256, 0, stream>>>(w_fc2, wf2h, n);
  }

  // 2) LN1: x -> h1 (f16)
  layernorm_kernel<<<MBT, 256, 0, stream>>>(x, ln1_g, ln1_b, h1, DIM);

  // 3) qkv = h1 @ w_qkv^T  (f16 out)
  gemm_nt_kernel<false, false, false, false, true>
      <<<dim3(cdiv(QKVN, 256), cdiv(MBT, 64), 1), 256, 0, stream>>>(
          h1, wqh, nullptr, nullptr, nullptr, qkvh,
          (int)MBT, QKVN, DIM, 0, 0, 0);

  // 4) split/pad q,k,v
  {
    long n = ZH * TP * HD;
    qkv_reshape_kernel<<<cdiv(n, 256), 256, 0, stream>>>(qkvh, Qb, Kb, Vb);
    v_transpose_kernel<<<cdiv(n, 256), 256, 0, stream>>>(Vb, Vt);
  }

  // 5) scores = q @ k^T  (batched over B*H)
  gemm_nt_kernel<false, false, false, true, false>
      <<<dim3(1, cdiv(TP, 64), (unsigned)ZH), 256, 0, stream>>>(
          Qb, Kb, nullptr, nullptr, Sc, nullptr,
          TP, TP, HD, (long)TP * HD, (long)TP * HD, (long)TP * TP);

  // 6) softmax (scale + log(size) bias, mask to T, zero pad)
  softmax_kernel<<<dim3(TP, (unsigned)ZH), 256, 0, stream>>>(Sc, asz, Pa);

  // 7) o = attn @ v  -> head-interleaved (B,T,DIM) f16
  attn_av_kernel<<<dim3(1, 1, (unsigned)ZH), 256, 0, stream>>>(Pa, Vt, Oh);

  // 8) x1 = x + o @ w_proj^T + b_proj  (f32)
  gemm_nt_kernel<true, true, false, true, false>
      <<<dim3(cdiv(DIM, 256), cdiv(MBT, 64), 1), 256, 0, stream>>>(
          Oh, wph, b_proj, x, x1, nullptr, (int)MBT, DIM, DIM, 0, 0, 0);

  // 9) metric = mean_h(k)
  {
    long n = MBT * HD;
    metric_kernel<<<cdiv(n, 256), 256, 0, stream>>>(Kb, metr);
  }

  // 10) bipartite soft matching
  matching_kernel<<<Bsz, 128, 0, stream>>>(metr, unm, srci, dsti);

  // 11) merge: gather unm + dst base rows (x1*size), write merged sizes
  merge_gather_kernel<<<dim3(TN, Bsz), 256, 0, stream>>>(x1, asz, unm, xs, out_sz);

  // 12) merge: scatter-add the R_EFF merged source rows
  merge_scatter_kernel<<<dim3(R_EFF, Bsz), 256, 0, stream>>>(x1, asz, srci, dsti,
                                                             xs, out_sz);

  // 13) x2 = xs / new_size
  {
    long n = MBN * DIM;
    normalize_kernel<<<cdiv(n, 256), 256, 0, stream>>>(xs, out_sz, x2);
  }

  // 14) LN2: x2 -> h2 (f16)
  layernorm_kernel<<<MBN, 256, 0, stream>>>(x2, ln2_g, ln2_b, h2, DIM);

  // 15) h3 = gelu(h2 @ w_fc1^T + b_fc1)  (f16)
  gemm_nt_kernel<true, false, true, false, true>
      <<<dim3(cdiv(MLP_H, 256), cdiv(MBN, 64), 1), 256, 0, stream>>>(
          h2, wf1h, b_fc1, nullptr, nullptr, h3, (int)MBN, MLP_H, DIM, 0, 0, 0);

  // 16) out_x = x2 + h3 @ w_fc2^T + b_fc2  (f32, final output)
  gemm_nt_kernel<true, true, false, true, false>
      <<<dim3(cdiv(DIM, 256), cdiv(MBN, 64), 1), 256, 0, stream>>>(
          h3, wf2h, b_fc2, x2, out_x, nullptr, (int)MBN, DIM, MLP_H, 0, 0, 0);

  // 17) rci from index sets
  rci_kernel<<<Bsz, 256, 0, stream>>>(unm, srci, dsti, out_rci);
}
